// SNN_11115375362658
// MI455X (gfx1250) — matrix operand order repaired and checked by reading, not compiled, not run on hardware
//
#include <hip/hip_runtime.h>
#include <hip/hip_bf16.h>

typedef __attribute__((ext_vector_type(16))) _Float16 v16h;
typedef __attribute__((ext_vector_type(8)))  float    v8f;
typedef __attribute__((ext_vector_type(4)))  unsigned v4u;
typedef __attribute__((ext_vector_type(8)))  int      v8i;
typedef __attribute__((ext_vector_type(4)))  int      v4i;

#define BB 64
#define TT 512
#define II 256
#define HH 512

// ---- WMMA 16-bit A-matrix (16xK tile, K-step=32) VGPR layout inverse maps ----
// Forward (contiguous k order used here, not the ISA table order): element j of v16h (v=j/2, h=j%2), lane L:
//   M = L%16 ; K = (L<16?0:16) + j
// Note: for even k, (k, k+1) share one lane and one fragment dword (v).
__device__ __forceinline__ int a_frag_lane(int mm, int k) {
    return mm + ((k >= 16) ? 16 : 0);
}
__device__ __forceinline__ int a_frag_j(int k) {
    return k & 15;
}
// B-matrix (32x16 f16): lane L -> N = L%16, element j -> K = (L<16?0:16) + j

// ---- TDM: async 2-D tile load (16 rows x 512 f32) global -> LDS ----
// D# per CDNA5 ISA. Groups 2/3 zero (<=2-D tensor).
__device__ __forceinline__ void tdm_load_z1_tile(const float* gsrc,
                                                 unsigned lds_off) {
    unsigned long long ga = (unsigned long long)(uintptr_t)gsrc;
    v4u g0;
    g0[0] = 1u;                                   // count=1 (valid), user mode
    g0[1] = lds_off;                              // lds_addr (bytes)
    g0[2] = (unsigned)(ga & 0xFFFFFFFFu);         // global_addr[31:0]
    g0[3] = (unsigned)((ga >> 32) & 0x01FFFFFFu)  // global_addr[56:32]
          | (2u << 30);                           // type = 2 ("image")
    v8i g1;
    g1[0] = (int)(2u << 16);                      // data_size=2 (4B), wg_mask=0
    g1[1] = (int)(512u << 16);                    // tensor_dim0 = 512
    g1[2] = (int)(16u << 16);                     // tensor_dim1 = 16
    g1[3] = (int)(512u << 16);                    // tile_dim0  = 512
    g1[4] = 16;                                   // tile_dim1  = 16
    g1[5] = (int)(TT * HH);                       // tensor_dim0_stride = 262144
    g1[6] = 0;
    g1[7] = 0;
    v4i g2 = {0, 0, 0, 0};
    v4i g3 = {0, 0, 0, 0};
#if defined(__clang_major__) && __clang_major__ >= 23
    v8i g4 = {0, 0, 0, 0, 0, 0, 0, 0};
    __builtin_amdgcn_tensor_load_to_lds(g0, g1, g2, g3, g4, 0);
#else
    __builtin_amdgcn_tensor_load_to_lds(g0, g1, g2, g3, 0);
#endif
}

// =====================================================================
// Phase 1: z1[bt][h] = x[bt][:] . W1[h][:] + b1[h]   (M=32768,N=512,K=256)
// WG: 256 thr (8 waves), tile M=128 x N=64. W1 slice staged once into LDS
// fragments (f32->f16), x tile staged per K-step. 256 WMMAs / WG.
// =====================================================================
__global__ __launch_bounds__(256)
void snn_z1_gemm(const float* __restrict__ x,
                 const float* __restrict__ conv1_w,
                 const float* __restrict__ conv1_b,
                 float* __restrict__ z1)
{
    __shared__ _Float16 shB[8 * 4 * 32 * 16];  // [kstep][ntile][lane][j] 32 KB
    __shared__ _Float16 shA[8 * 32 * 16];      // [mtile][lane][j] (one K-step) 8 KB

    const int tid  = threadIdx.x;
    const int wave = tid >> 5;
    const int lane = tid & 31;
    const int bt0  = blockIdx.x * 128;
    const int n0   = blockIdx.y * 64;

    // Stage B = W1^T slice as f16 fragments (middle tap of conv1_w, stride 3)
    for (int idx = tid; idx < 8 * 4 * 32 * 16; idx += 256) {
        int j     = idx & 15;
        int ln    = (idx >> 4) & 31;
        int ntile = (idx >> 9) & 3;
        int kstep = idx >> 11;
        int n = n0 + ntile * 16 + (ln & 15);
        int k = kstep * 32 + ((ln < 16) ? 0 : 16) + j;
        shB[idx] = (_Float16)conv1_w[(size_t)(n * II + k) * 3 + 1];
    }

    v8f c[4] = {};
    for (int ks = 0; ks < 8; ++ks) {
        // Stage A tile (128 rows x 32 cols f32 -> packed-f16 fragment dwords)
        {
            int m  = tid >> 1;
            int c0 = (tid & 1) * 16;
            const float* src = x + (size_t)(bt0 + m) * II + ks * 32 + c0;
            int mtile = m >> 4, mm = m & 15;
            unsigned* shA32 = (unsigned*)shA;
            #pragma unroll
            for (int e = 0; e < 8; ++e) {
                int k = c0 + 2 * e;                 // even k: pair in one dword
                _Float16 h0 = (_Float16)src[2 * e];
                _Float16 h1 = (_Float16)src[2 * e + 1];
                unsigned pk = (unsigned)__builtin_bit_cast(unsigned short, h0)
                            | ((unsigned)__builtin_bit_cast(unsigned short, h1) << 16);
                shA32[(mtile * 32 + a_frag_lane(mm, k)) * 8 + (a_frag_j(k) >> 1)] = pk;
            }
        }
        __syncthreads();
        v16h a = *(const v16h*)&shA[(wave * 32 + lane) * 16];
        #pragma unroll
        for (int nt = 0; nt < 4; ++nt) {
            v16h b = *(const v16h*)&shB[((ks * 4 + nt) * 32 + lane) * 16];
            c[nt] = __builtin_amdgcn_wmma_f32_16x16x32_f16(
                        false, a, false, b, (short)0, c[nt], false, false);
        }
        __syncthreads();
    }

    // bias + store (C layout: VGPR r -> M = r (lanes 0-15) / 8+r (lanes 16-31))
    #pragma unroll
    for (int nt = 0; nt < 4; ++nt) {
        int n = n0 + nt * 16 + (lane & 15);
        float bias = conv1_b[n];
        int mrow0 = bt0 + wave * 16 + ((lane < 16) ? 0 : 8);
        #pragma unroll
        for (int r = 0; r < 8; ++r)
            z1[(size_t)(mrow0 + r) * HH + n] = c[nt][r] + bias;
    }
}

// =====================================================================
// Phase 2: LIF scan over T. Grid = 4 b-tiles x 8 N-slices = 32 WGs of 128 thr.
// m2 lives in WMMA accumulators across all 512 steps; W2^T fragments held in
// 128 VGPRs/lane (loaded once); s1 fragments rebuilt in LDS each step.
// z1 tiles (16x512 f32, 32 KB) streamed in by the Tensor Data Mover with
// double buffering, overlapping the serial WMMA chain of the scan.
// =====================================================================
__global__ __launch_bounds__(128)
void snn_scan(const float* __restrict__ z1,
              const float* __restrict__ conv2_w,
              const float* __restrict__ conv2_b,
              const float* __restrict__ th1p,
              const float* __restrict__ th2p,
              float* __restrict__ out)
{
    __shared__ float    shZ[2][16 * 512];    // double-buffered z1 tiles, 64 KB
    __shared__ _Float16 shS1[16 * 32 * 16];  // [kstep][lane][j] 16 KB

    const int tid  = threadIdx.x;
    const int wave = tid >> 5;
    const int lane = tid & 31;
    const int b0 = (blockIdx.x >> 3) * 16;
    const int n0 = (blockIdx.x & 7) * 64;

    const float th1 = th1p[0], th2 = th2p[0];
    const float inv_th1 = 1.0f / th1, inv_th2 = 1.0f / th2;

    // Preload this wave's B fragments: W2^T[k][n], n = n0+wave*16+lane%16
    v16h bfr[16];
    {
        int n   = n0 + wave * 16 + (lane & 15);
        int khi = (lane < 16) ? 0 : 16;
        #pragma unroll
        for (int ks = 0; ks < 16; ++ks)
            #pragma unroll
            for (int j = 0; j < 16; ++j)
                bfr[ks][j] = (_Float16)
                    conv2_w[((size_t)n * HH + ks * 32 + khi + j) * 3 + 1];
    }
    const float bias2 = conv2_b[n0 + wave * 16 + (lane & 15)];

    float m1r[64];
    #pragma unroll
    for (int j = 0; j < 64; ++j) m1r[j] = 0.0f;
    v8f m2 = {};  // persistent layer-2 membrane

    const unsigned ldsZ0 = (unsigned)(unsigned long long)(uintptr_t)&shZ[0][0];
    const unsigned ldsZ1 = (unsigned)(unsigned long long)(uintptr_t)&shZ[1][0];

    // Kick off TDM load of the t=0 tile into buffer 0 (wave 0 only; TDM
    // ignores EXEC, one issue per workgroup).
    if (tid < 32)
        tdm_load_z1_tile(z1 + (size_t)b0 * TT * HH, ldsZ0);

    unsigned* shS32 = (unsigned*)shS1;

    for (int t = 0; t < TT; ++t) {
        const int buf = t & 1;
        // Prefetch tile t+1 into the other buffer, then guarantee tile t is
        // resident (TDM completes in order -> TENSORcnt<=1 means t is done).
        if (tid < 32) {
            if (t + 1 < TT) {
                tdm_load_z1_tile(z1 + ((size_t)b0 * TT + (t + 1)) * HH,
                                 buf ? ldsZ0 : ldsZ1);
                __builtin_amdgcn_s_wait_tensorcnt((short)1);
            } else {
                __builtin_amdgcn_s_wait_tensorcnt((short)0);
            }
        }
        __syncthreads();

        // ---- layer-1 membrane update + spike, packed s1 fragment stores ----
        // Each thread owns 32 adjacent (h1, h1+1) pairs: one ds_load_b64 of
        // z1 and one packed ds_store_b32 of the f16 spike pair (no v_cvt:
        // spikes are 0x3C00 / 0x0000 by construction).
        #pragma unroll
        for (int p = 0; p < 32; ++p) {
            int e  = 2 * tid + 256 * p;   // even; e = mm*512 + h1
            int mm = e >> 9, h1 = e & 511;
            float2 zz = *(const float2*)&shZ[buf][e];
            float a0 = m1r[2 * p]     + zz.x;
            float a1 = m1r[2 * p + 1] + zz.y;
            float t0 = a0 * inv_th1 - 1.0f;
            float t1 = a1 * inv_th1 - 1.0f;
            unsigned pk = ((t0 >= 0.0f) ? 0x00003C00u : 0u)
                        | ((t1 >= 0.0f) ? 0x3C000000u : 0u);
            m1r[2 * p]     = a0 - ((t0 > 0.0f) ? th1 : 0.0f);
            m1r[2 * p + 1] = a1 - ((t1 > 0.0f) ? th1 : 0.0f);
            int ks = h1 >> 5, kk = h1 & 31;  // kk even -> pair shares dword
            shS32[(ks * 32 + a_frag_lane(mm, kk)) * 8 + (a_frag_j(kk) >> 1)] = pk;
        }
        __syncthreads();

        // ---- m2 += s1 @ W2^T  (16 chained WMMAs, K=512) ----
        #pragma unroll
        for (int ks = 0; ks < 16; ++ks) {
            v16h a = *(const v16h*)&shS1[(ks * 32 + lane) * 16];
            m2 = __builtin_amdgcn_wmma_f32_16x16x32_f16(
                     false, a, false, bfr[ks], (short)0, m2, false, false);
        }

        // ---- bias, spike, reset, emit s2 ----
        {
            int n = n0 + wave * 16 + (lane & 15);
            #pragma unroll
            for (int r = 0; r < 8; ++r) {
                int brow = b0 + ((lane < 16) ? r : 8 + r);
                float m2v = m2[r] + bias2;
                float thr = m2v * inv_th2 - 1.0f;
                float s2  = (thr >= 0.0f) ? 1.0f : 0.0f;
                m2[r] = m2v - ((thr > 0.0f) ? th2 : 0.0f);
                out[((size_t)brow * TT + t) * HH + n] = s2;
            }
        }
        __syncthreads();  // protect shS1 (and z1 buffer reuse) for next step
    }
}

extern "C" void kernel_launch(void* const* d_in, const int* in_sizes, int n_in,
                              void* d_out, int out_size, void* d_ws, size_t ws_size,
                              hipStream_t stream) {
    const float* x       = (const float*)d_in[0];
    const float* conv1_w = (const float*)d_in[1];
    const float* conv1_b = (const float*)d_in[2];
    const float* conv2_w = (const float*)d_in[3];
    const float* conv2_b = (const float*)d_in[4];
    const float* th1     = (const float*)d_in[5];
    const float* th2     = (const float*)d_in[6];
    float* out = (float*)d_out;
    float* z1  = (float*)d_ws;  // needs 64*512*512*4 = 64 MB

    dim3 g1((BB * TT) / 128, HH / 64);   // 256 x 8 WGs
    snn_z1_gemm<<<g1, 256, 0, stream>>>(x, conv1_w, conv1_b, z1);

    snn_scan<<<32, 128, 0, stream>>>(z1, conv2_w, conv2_b, th1, th2, out);
}
